// RecurrentRGCN_29317446763354
// MI455X (gfx1250) — matrix operand order repaired and checked by reading, not compiled, not run on hardware
//
#include <hip/hip_runtime.h>
#include <hip/hip_bf16.h>

// ---------------- problem constants (match reference) ----------------
constexpr int HD  = 256;      // hidden dim
constexpr int R2C = 400;      // num relations*2
constexpr int NE  = 30000;    // num entities
constexpr int TS  = 8;        // time steps
constexpr int EC  = 250000;   // edges per step
constexpr float SLOPE = 0.22916666666666666f; // rrelu eval slope

typedef __attribute__((ext_vector_type(16))) __bf16 v16bf;
typedef __attribute__((ext_vector_type(8)))  __bf16 v8bf;
typedef __attribute__((ext_vector_type(8)))  float  v8f;

__device__ __forceinline__ float sigmoidf_(float x) { return 1.0f / (1.0f + __expf(-x)); }

// ---------------- generic WMMA GEMM: C[MxNc] = A[MxK] @ op(B) ----------------
// transB==0: B is [K x Nc] row major; transB==1: B is [Nc x K] (x @ W.T).
// Block = 256 threads (8 waves). Macro-tile 16(M) x 256(N), K-step 32.
// Staging: issue all global loads into regs first, then convert+ds_store
// (load clause + incremental loadcnt drain instead of per-load stalls).
__global__ void wmma_gemm_bf16(const float* __restrict__ A, int lda,
                               const float* __restrict__ B, int ldb,
                               int transB,
                               float* __restrict__ C, int ldc, int K) {
  __shared__ __align__(64) __bf16 lA[16][32];    // [m][k]   1 KB
  __shared__ __align__(64) __bf16 lBT[256][32];  // [n][k]  16 KB
  const int tid  = threadIdx.x;
  const int wv   = tid >> 5;
  const int lane = tid & 31;
  const int half = lane >> 4;
  const int l15  = lane & 15;
  const int m0 = blockIdx.x << 4;
  const int n0 = blockIdx.y << 8;

  v8f acc0 = {0.f, 0.f, 0.f, 0.f, 0.f, 0.f, 0.f, 0.f};
  v8f acc1 = {0.f, 0.f, 0.f, 0.f, 0.f, 0.f, 0.f, 0.f};
  const int nn0 = (wv << 5) + l15;

  for (int k0 = 0; k0 < K; k0 += 32) {
    __syncthreads();
    // ---- load phase: A (float2) + B (8x float4) into registers ----
    const int aidx = tid << 1;
    const int ar = aidx >> 5, ac = aidx & 31;
    const float2 av = *(const float2*)(A + (size_t)(m0 + ar) * lda + (k0 + ac));
    float4 t[8];
    if (!transB) {
#pragma unroll
      for (int i = 0; i < 8; ++i) {
        const int s  = tid + (i << 8);
        const int r  = s >> 6;
        const int c4 = (s & 63) << 2;
        t[i] = *(const float4*)(B + (size_t)(k0 + r) * ldb + (n0 + c4));
      }
    } else {
#pragma unroll
      for (int i = 0; i < 8; ++i) {
        const int s  = tid + (i << 8);
        const int n  = s >> 3;
        const int k4 = (s & 7) << 2;
        t[i] = *(const float4*)(B + (size_t)(n0 + n) * ldb + (k0 + k4));
      }
    }
    if (k0 + 32 < K && tid < 16)
      __builtin_prefetch(A + (size_t)(m0 + tid) * lda + (k0 + 32), 0, 1);
    // ---- store phase: convert to bf16 and scatter into LDS ----
    lA[ar][ac]     = (__bf16)av.x;
    lA[ar][ac + 1] = (__bf16)av.y;
    if (!transB) {
#pragma unroll
      for (int i = 0; i < 8; ++i) {
        const int s  = tid + (i << 8);
        const int r  = s >> 6;
        const int c4 = (s & 63) << 2;
        lBT[c4 + 0][r] = (__bf16)t[i].x;
        lBT[c4 + 1][r] = (__bf16)t[i].y;
        lBT[c4 + 2][r] = (__bf16)t[i].z;
        lBT[c4 + 3][r] = (__bf16)t[i].w;
      }
    } else {
#pragma unroll
      for (int i = 0; i < 8; ++i) {
        const int s  = tid + (i << 8);
        const int n  = s >> 3;
        const int k4 = (s & 7) << 2;
        lBT[n][k4 + 0] = (__bf16)t[i].x;
        lBT[n][k4 + 1] = (__bf16)t[i].y;
        lBT[n][k4 + 2] = (__bf16)t[i].z;
        lBT[n][k4 + 3] = (__bf16)t[i].w;
      }
    }
    __syncthreads();
    // ---- fragments (contiguous k = 16*half + e for both A and B) ----
    v16bf a;
    {
      const v8bf alo = *(const v8bf*)&lA[l15][half * 16];
      const v8bf ahi = *(const v8bf*)&lA[l15][half * 16 + 8];
#pragma unroll
      for (int e = 0; e < 8; ++e) { a[e] = alo[e]; a[e + 8] = ahi[e]; }
    }
    const v16bf b0 = *(const v16bf*)&lBT[nn0][half * 16];
    const v16bf b1 = *(const v16bf*)&lBT[nn0 + 16][half * 16];
    acc0 = __builtin_amdgcn_wmma_f32_16x16x32_bf16(false, a, false, b0,
                                                   (short)0, acc0, false, false);
    acc1 = __builtin_amdgcn_wmma_f32_16x16x32_bf16(false, a, false, b1,
                                                   (short)0, acc1, false, false);
  }
  const int cn = n0 + (wv << 5) + l15;
#pragma unroll
  for (int v = 0; v < 8; ++v) {
    const size_t row = (size_t)(m0 + v + half * 8) * ldc;
    C[row + cn]      = acc0[v];
    C[row + cn + 16] = acc1[v];
  }
}

// ---------------- fused quad GEMM: Cw = A @ Bw for w=0..3 ----------------
// A: [M x 256]; Bw: [256 x 256] row major; Cw: [M x 256]. Stages A once per
// K-step and keeps 4 B tiles in LDS (64KB) -> 8 WMMAs per wave per K-step,
// and h is read from HBM once instead of 4x.
__global__ void wmma_gemm_bf16_quad(const float* __restrict__ A,
                                    const float* __restrict__ Bw0,
                                    const float* __restrict__ Bw1,
                                    const float* __restrict__ Bw2,
                                    const float* __restrict__ Bw3,
                                    float* __restrict__ C0, float* __restrict__ C1,
                                    float* __restrict__ C2, float* __restrict__ C3) {
  __shared__ __align__(64) __bf16 lA[16][32];        //  1 KB
  __shared__ __align__(64) __bf16 lBT[4][256][32];   // 64 KB
  const int tid  = threadIdx.x;
  const int wv   = tid >> 5;
  const int lane = tid & 31;
  const int half = lane >> 4;
  const int l15  = lane & 15;
  const int m0 = blockIdx.x << 4;
  const float* Bs[4] = {Bw0, Bw1, Bw2, Bw3};

  const v8f z8 = {0.f, 0.f, 0.f, 0.f, 0.f, 0.f, 0.f, 0.f};
  v8f acc[4][2];
#pragma unroll
  for (int w = 0; w < 4; ++w) { acc[w][0] = z8; acc[w][1] = z8; }
  const int nn0 = (wv << 5) + l15;

  for (int k0 = 0; k0 < HD; k0 += 32) {
    __syncthreads();
    const int aidx = tid << 1;
    const int ar = aidx >> 5, ac = aidx & 31;
    const float2 av = *(const float2*)(A + (size_t)(m0 + ar) * HD + (k0 + ac));
    lA[ar][ac]     = (__bf16)av.x;
    lA[ar][ac + 1] = (__bf16)av.y;
    if (k0 + 32 < HD && tid < 16)
      __builtin_prefetch(A + (size_t)(m0 + tid) * HD + (k0 + 32), 0, 1);
#pragma unroll
    for (int w = 0; w < 4; ++w) {
      float4 t[8];
#pragma unroll
      for (int i = 0; i < 8; ++i) {
        const int s  = tid + (i << 8);
        const int r  = s >> 6;
        const int c4 = (s & 63) << 2;
        t[i] = *(const float4*)(Bs[w] + (size_t)(k0 + r) * HD + c4);
      }
#pragma unroll
      for (int i = 0; i < 8; ++i) {
        const int s  = tid + (i << 8);
        const int r  = s >> 6;
        const int c4 = (s & 63) << 2;
        lBT[w][c4 + 0][r] = (__bf16)t[i].x;
        lBT[w][c4 + 1][r] = (__bf16)t[i].y;
        lBT[w][c4 + 2][r] = (__bf16)t[i].z;
        lBT[w][c4 + 3][r] = (__bf16)t[i].w;
      }
    }
    __syncthreads();
    v16bf a;
    {
      const v8bf alo = *(const v8bf*)&lA[l15][half * 16];
      const v8bf ahi = *(const v8bf*)&lA[l15][half * 16 + 8];
#pragma unroll
      for (int e = 0; e < 8; ++e) { a[e] = alo[e]; a[e + 8] = ahi[e]; }
    }
#pragma unroll
    for (int w = 0; w < 4; ++w) {
      const v16bf b0 = *(const v16bf*)&lBT[w][nn0][half * 16];
      const v16bf b1 = *(const v16bf*)&lBT[w][nn0 + 16][half * 16];
      acc[w][0] = __builtin_amdgcn_wmma_f32_16x16x32_bf16(false, a, false, b0,
                                                          (short)0, acc[w][0], false, false);
      acc[w][1] = __builtin_amdgcn_wmma_f32_16x16x32_bf16(false, a, false, b1,
                                                          (short)0, acc[w][1], false, false);
    }
  }
  float* Cs[4] = {C0, C1, C2, C3};
  const int cn = (wv << 5) + l15;
#pragma unroll
  for (int w = 0; w < 4; ++w) {
#pragma unroll
    for (int v = 0; v < 8; ++v) {
      const size_t row = (size_t)(m0 + v + half * 8) * HD;
      Cs[w][row + cn]      = acc[w][0][v];
      Cs[w][row + cn + 16] = acc[w][1][v];
    }
  }
}

// ---------------- helpers ----------------
__global__ void zero_f32(float* __restrict__ p, long n) {
  long i = (long)blockIdx.x * blockDim.x + threadIdx.x;
  long s = (long)gridDim.x * blockDim.x;
  for (; i < n; i += s) p[i] = 0.0f;
}

__global__ void copy_f32(const float* __restrict__ src, float* __restrict__ dst, long n) {
  long i = (long)blockIdx.x * blockDim.x + threadIdx.x;
  long s = (long)gridDim.x * blockDim.x;
  for (; i < n; i += s) dst[i] = src[i];
}

// per-relation segment sum of h[rte] (one block per edge, HD threads)
__global__ void seg_gather(const float* __restrict__ h,
                           const int* __restrict__ ent,
                           const int* __restrict__ rel,
                           float* __restrict__ sums, float* __restrict__ cnts) {
  const int e = blockIdx.x;
  const int j = threadIdx.x;
  const int r = rel[e];
  const int v = ent[e];
  atomicAdd(&sums[(size_t)r * HD + j], h[(size_t)v * HD + j]);
  if (j == 0) atomicAdd(&cnts[r], 1.0f);
}

__global__ void seg_mean(const float* __restrict__ sums, const float* __restrict__ cnts,
                         float* __restrict__ mean) {
  const int idx = blockIdx.x * blockDim.x + threadIdx.x;   // R2C*HD
  const float c = cnts[idx / HD];
  mean[idx] = sums[idx] / fmaxf(c, 1.0f);
}

__global__ void concat_x(const float* __restrict__ emb_rel, const float* __restrict__ xmean,
                         float* __restrict__ xin) {
  const int idx = blockIdx.x * blockDim.x + threadIdx.x;   // R2C*2*HD
  const int row = idx / (2 * HD);
  const int col = idx - row * 2 * HD;
  xin[idx] = (col < HD) ? emb_rel[(size_t)row * HD + col]
                        : xmean[(size_t)row * HD + (col - HD)];
}

// GRUCell gates (r,z,n ordering, torch semantics)
__global__ void gru_elem(const float* __restrict__ gi, const float* __restrict__ gh,
                         const float* __restrict__ bih, const float* __restrict__ bhh,
                         const float* __restrict__ h0, float* __restrict__ h0n) {
  const int idx = blockIdx.x * blockDim.x + threadIdx.x;   // R2C*HD
  const int i = idx / HD;
  const int j = idx - i * HD;
  const float* gir = gi + (size_t)i * 3 * HD;
  const float* ghr = gh + (size_t)i * 3 * HD;
  const float r  = sigmoidf_(gir[j] + bih[j] + ghr[j] + bhh[j]);
  const float z  = sigmoidf_(gir[HD + j] + bih[HD + j] + ghr[HD + j] + bhh[HD + j]);
  const float nn = tanhf(gir[2 * HD + j] + bih[2 * HD + j] +
                         r * (ghr[2 * HD + j] + bhh[2 * HD + j]));
  h0n[idx] = (1.0f - z) * nn + z * h0[idx];
}

// msg = hW[src] + rW[etype]; scatter-add by dst (one block per edge)
__global__ void edge_scatter(const int* __restrict__ src, const int* __restrict__ dst,
                             const int* __restrict__ etype,
                             const float* __restrict__ hW, const float* __restrict__ rW,
                             float* __restrict__ agg, float* __restrict__ indeg) {
  const int e = blockIdx.x;
  const int j = threadIdx.x;
  const int s = src[e], d = dst[e], r = etype[e];
  const float val = hW[(size_t)s * HD + j] + rW[(size_t)r * HD + j];
  atomicAdd(&agg[(size_t)d * HD + j], val);
  if (j == 0) atomicAdd(&indeg[d], 1.0f);
}

// node update: normalize agg, select loop weight path, rrelu, time gate
__global__ void node_update(const float* __restrict__ agg, const float* __restrict__ indeg,
                            const float* __restrict__ hL, const float* __restrict__ hE,
                            const float* __restrict__ hT, const float* __restrict__ tgb,
                            const float* __restrict__ h, float* __restrict__ out) {
  const int idx = blockIdx.x * blockDim.x + threadIdx.x;   // NE*HD
  const int i = idx / HD;
  const int j = idx - i * HD;
  const float ind = indeg[i];
  const float a = agg[idx] / fmaxf(ind, 1.0f);
  const float loop = (ind > 0.0f) ? hL[idx] : hE[idx];
  const float nr = a + loop;
  const float cur = (nr >= 0.0f) ? nr : nr * SLOPE;
  const float tw = sigmoidf_(hT[idx] + tgb[j]);
  out[idx] = tw * cur + (1.0f - tw) * h[idx];
}

// ---------------- driver ----------------
extern "C" void kernel_launch(void* const* d_in, const int* in_sizes, int n_in,
                              void* d_out, int out_size, void* d_ws, size_t ws_size,
                              hipStream_t stream) {
  const float* dyn     = (const float*)d_in[0];
  const float* emb_rel = (const float*)d_in[1];
  const float* w_ih    = (const float*)d_in[2];   // [3H, 2H]
  const float* w_hh    = (const float*)d_in[3];   // [3H, H]
  const float* b_ih    = (const float*)d_in[4];
  const float* b_hh    = (const float*)d_in[5];
  const float* w_nb    = (const float*)d_in[6];   // [H, H]
  const float* w_loop  = (const float*)d_in[7];
  const float* w_evo   = (const float*)d_in[8];
  const float* w_tg    = (const float*)d_in[9];
  const float* b_tg    = (const float*)d_in[10];
  const int* e_src  = (const int*)d_in[11];
  const int* e_dst  = (const int*)d_in[12];
  const int* e_type = (const int*)d_in[13];
  const int* rte    = (const int*)d_in[14];
  const int* rtrel  = (const int*)d_in[15];

  float* out = (float*)d_out;   // history [TS*NE*HD] then h0_fin [R2C*HD]
  float* W   = (float*)d_ws;

  size_t off = 0;
  float* x_sums = W + off; off += (size_t)R2C * HD;
  float* x_cnts = W + off; off += R2C;
  float* x_mean = W + off; off += (size_t)R2C * HD;
  float* x_in   = W + off; off += (size_t)R2C * 2 * HD;
  float* gi     = W + off; off += (size_t)R2C * 3 * HD;
  float* gh     = W + off; off += (size_t)R2C * 3 * HD;
  float* h0a    = W + off; off += (size_t)R2C * HD;
  float* h0b    = W + off; off += (size_t)R2C * HD;
  float* rW     = W + off; off += (size_t)R2C * HD;
  float* hW     = W + off; off += (size_t)NE * HD;
  float* hL     = W + off; off += (size_t)NE * HD;
  float* hE     = W + off; off += (size_t)NE * HD;
  float* hT     = W + off; off += (size_t)NE * HD;
  float* agg    = W + off; off += (size_t)NE * HD;
  float* indeg  = W + off; off += NE;
  (void)ws_size; (void)in_sizes; (void)n_in; (void)out_size;

  const dim3 blkG(256), blk256(256);
  const dim3 gemmR2_3H(R2C / 16, 3 * HD / 256);  // 25 x 3
  const dim3 gemmR2_H (R2C / 16, HD / 256);      // 25 x 1
  const dim3 quadN    (NE / 16, 1);              // 1875

  for (int t = 0; t < TS; ++t) {
    const float* h   = (t == 0) ? dyn : out + (size_t)(t - 1) * NE * HD;
    const float* h0i = (t == 0) ? emb_rel : ((t & 1) ? h0a : h0b); // h0_out(t-1)
    float*       h0o = (t & 1) ? h0b : h0a;

    // --- per-relation segment mean of h[rte] ---
    zero_f32<<<2048, blk256, 0, stream>>>(x_sums, (long)R2C * HD + R2C);
    seg_gather<<<EC, HD, 0, stream>>>(h, rte + (size_t)t * EC, rtrel + (size_t)t * EC,
                                      x_sums, x_cnts);
    seg_mean<<<(R2C * HD) / 256, blk256, 0, stream>>>(x_sums, x_cnts, x_mean);

    // --- GRU cell: gi = [emb_rel|x_mean] @ w_ih^T ; gh = h0 @ w_hh^T ---
    concat_x<<<(R2C * 2 * HD) / 256, blk256, 0, stream>>>(emb_rel, x_mean, x_in);
    wmma_gemm_bf16<<<gemmR2_3H, blkG, 0, stream>>>(x_in, 2 * HD, w_ih, 2 * HD, 1,
                                                   gi, 3 * HD, 2 * HD);
    wmma_gemm_bf16<<<gemmR2_3H, blkG, 0, stream>>>(h0i, HD, w_hh, HD, 1,
                                                   gh, 3 * HD, HD);
    gru_elem<<<(R2C * HD) / 256, blk256, 0, stream>>>(gi, gh, b_ih, b_hh, h0i, h0o);

    // --- dense GEMMs: (h[src]+h0[etype])@W -> (h@W)[src] + (h0@W)[etype] ---
    wmma_gemm_bf16<<<gemmR2_H, blkG, 0, stream>>>(h0o, HD, w_nb, HD, 0, rW, HD, HD);
    // fused: hW/hL/hE/hT all from one pass over h
    wmma_gemm_bf16_quad<<<quadN, blkG, 0, stream>>>(h, w_nb, w_loop, w_evo, w_tg,
                                                    hW, hL, hE, hT);

    // --- message passing scatter + node update ---
    zero_f32<<<8192, blk256, 0, stream>>>(agg, (long)NE * HD + NE);
    edge_scatter<<<EC, HD, 0, stream>>>(e_src + (size_t)t * EC, e_dst + (size_t)t * EC,
                                        e_type + (size_t)t * EC, hW, rW, agg, indeg);
    node_update<<<(NE * HD) / 256, blk256, 0, stream>>>(agg, indeg, hL, hE, hT, b_tg,
                                                        h, out + (size_t)t * NE * HD);
  }

  // h0 after step 7 lives in h0b (t=7 odd); append as h0_fin
  copy_f32<<<400, blk256, 0, stream>>>(h0b, out + (size_t)TS * NE * HD, (long)R2C * HD);
}
